// DecoderRNN_33947421507826
// MI455X (gfx1250) — matrix operand order repaired and checked by reading, not compiled, not run on hardware
//
#include <hip/hip_runtime.h>

// DecoderRNN (attention LSTM, B=64,T=24,E=512,H=1024,A=1024,V=32000,L=25)
// bf16 WMMA (v_wmma_f32_16x16x32_bf16, fp32 accumulate) for all GEMMs.
// Weights packed once into WMMA fragment order; per-step GEMMs use an M4
// kernel (M=64 -> 4 tiles per wave) so streamed weight fragments are loaded
// exactly once. Activation bf16 packing is fused into producer kernels.

typedef __attribute__((ext_vector_type(16))) __bf16 v16bf;
typedef __attribute__((ext_vector_type(8)))  float  v8f;
typedef unsigned short u16; // bf16 bit container (host-safe)

#define DEV_INLINE __device__ __forceinline__

// ---------------------------------------------------------------------------
// Fragment index maps (CDNA5 ISA 7.12.2, wave32):
// A (16x32 MxK bf16): lane l: m=l&15, half=l>>4;
//   elem e: k = ((e&8)?16:0) + ((e>>1)&3)*2 + half*8 + (e&1)
// B (32x16 KxN bf16): lane l: n=l&15, half=l>>4; elem e: k = ((e&8)?16:0) + half*8 + (e&7)
// C/D (16x16 f32): vgpr r: m = r + half*8, n = l&15
// Packed: Apk[((mt*KS+ks)*32+lane)*16+e], Bpk[((nt*KS+ks)*32+lane)*16+e]
// ---------------------------------------------------------------------------

DEV_INLINE size_t apk_idx(int m, int k, int KS) {
  // inverse of the A-fragment map: (m,k) -> packed element index
  int mt = m >> 4, lo = m & 15;
  int ks = k >> 5, kk = k & 31;
  int half = (kk >> 3) & 1;
  int e = ((kk >> 4) << 3) | (kk & 7);
  return ((size_t)(mt * KS + ks) * 32 + half * 16 + lo) * 16 + e;
}

DEV_INLINE u16 f2bf(float v) {
  __bf16 b = (__bf16)v;
  return __builtin_bit_cast(u16, b);
}

__global__ void pack_a_bf16(const float* __restrict__ src, u16* __restrict__ dst, int KS) {
  int idx  = blockIdx.x * blockDim.x + threadIdx.x;
  int e    = idx & 15;
  int lane = (idx >> 4) & 31;
  int rem  = idx >> 9;
  int ks   = rem % KS, mt = rem / KS;
  int half = lane >> 4, lo = lane & 15;
  int K    = KS << 5;
  int m    = mt * 16 + lo;
  int k    = ks * 32 + ((e & 8) ? 16 : 0) + (((e >> 1) & 3) << 1) + (half << 3) + (e & 1);
  dst[idx] = f2bf(src[(size_t)m * K + k]);
}

__global__ void pack_b_bf16(const float* __restrict__ src, u16* __restrict__ dst,
                            int KS, int N) {
  int idx  = blockIdx.x * blockDim.x + threadIdx.x;
  int e    = idx & 15;
  int lane = (idx >> 4) & 31;
  int rem  = idx >> 9;
  int ks   = rem % KS, nt = rem / KS;
  int half = lane >> 4, lo = lane & 15;
  int k    = ks * 32 + ((e & 8) ? 16 : 0) + half * 8 + (e & 7);
  int n    = nt * 16 + lo;
  dst[idx] = f2bf(src[(size_t)k * N + n]);
}

__global__ void concat_bias(const float* __restrict__ b0, const float* __restrict__ b1,
                            float* __restrict__ dst, int n0) {
  int i = blockIdx.x * blockDim.x + threadIdx.x; // n0 + n1 threads
  dst[i] = (i < n0) ? b0[i] : b1[i - n0];
}

// generic: C[m,n] = A@B + bias (used once for embeds@We, M=1600)
__global__ __launch_bounds__(256)
void wmma_gemm_bf16(const u16* __restrict__ Apk, const u16* __restrict__ Bpk,
                    const float* __restrict__ bias, float* __restrict__ C,
                    int MT, int NT, int KS, int ldC) {
  int wave = (blockIdx.x * 256 + threadIdx.x) >> 5;
  int lane = threadIdx.x & 31;
  if (wave >= MT * NT) return;
  int mt = wave % MT, nt = wave / MT;
  const u16* ap = Apk + ((size_t)mt * KS * 32 + lane) * 16;
  const u16* bp = Bpk + ((size_t)nt * KS * 32 + lane) * 16;
  v8f acc = {};
  for (int ks = 0; ks < KS; ++ks) {
    v16bf a = *(const v16bf*)ap; ap += 512;
    v16bf b = *(const v16bf*)bp; bp += 512;
    acc = __builtin_amdgcn_wmma_f32_16x16x32_bf16(false, a, false, b, (short)0, acc,
                                                  false, false);
  }
  int lo = lane & 15, half = lane >> 4;
  int n  = nt * 16 + lo;
  float bv = bias[n];
  #pragma unroll
  for (int r = 0; r < 8; ++r)
    C[(size_t)(mt * 16 + half * 8 + r) * ldC + n] = acc[r] + bv;
}

// M=64 specialization: one wave = full 64-row column of one N tile.
// B fragment loaded once, used by 4 independent WMMAs (hides D->C latency).
__global__ __launch_bounds__(256)
void wmma_gemm_m4(const u16* __restrict__ Apk, const u16* __restrict__ Bpk,
                  const float* __restrict__ bias, float* __restrict__ C,
                  int NT, int KS, int ldC) {
  int nt   = (blockIdx.x * 256 + threadIdx.x) >> 5;
  int lane = threadIdx.x & 31;
  if (nt >= NT) return;
  const u16* bp = Bpk + ((size_t)nt * KS * 32 + lane) * 16;
  const u16* ap = Apk + (size_t)lane * 16;
  const size_t as = (size_t)KS * 512; // A stride per M tile
  v8f acc0 = {}, acc1 = {}, acc2 = {}, acc3 = {};
  for (int ks = 0; ks < KS; ++ks) {
    v16bf b  = *(const v16bf*)bp; bp += 512;
    v16bf a0 = *(const v16bf*)(ap);
    v16bf a1 = *(const v16bf*)(ap + as);
    v16bf a2 = *(const v16bf*)(ap + 2 * as);
    v16bf a3 = *(const v16bf*)(ap + 3 * as);
    ap += 512;
    acc0 = __builtin_amdgcn_wmma_f32_16x16x32_bf16(false, a0, false, b, (short)0, acc0, false, false);
    acc1 = __builtin_amdgcn_wmma_f32_16x16x32_bf16(false, a1, false, b, (short)0, acc1, false, false);
    acc2 = __builtin_amdgcn_wmma_f32_16x16x32_bf16(false, a2, false, b, (short)0, acc2, false, false);
    acc3 = __builtin_amdgcn_wmma_f32_16x16x32_bf16(false, a3, false, b, (short)0, acc3, false, false);
  }
  int lo = lane & 15, half = lane >> 4;
  int n  = nt * 16 + lo;
  float bv = bias[n];
  #pragma unroll
  for (int r = 0; r < 8; ++r) {
    int m = half * 8 + r;
    C[(size_t)(m)      * ldC + n] = acc0[r] + bv;
    C[(size_t)(m + 16) * ldC + n] = acc1[r] + bv;
    C[(size_t)(m + 32) * ldC + n] = acc2[r] + bv;
    C[(size_t)(m + 48) * ldC + n] = acc3[r] + bv;
  }
}

// M=64, C = A1@B1 + A2@B2 + bias : fused x@Wi + h@Wh
__global__ __launch_bounds__(256)
void wmma_gemm2_m4(const u16* __restrict__ A1, const u16* __restrict__ B1,
                   const u16* __restrict__ A2, const u16* __restrict__ B2,
                   const float* __restrict__ bias, float* __restrict__ C,
                   int NT, int KS, int ldC) {
  int nt   = (blockIdx.x * 256 + threadIdx.x) >> 5;
  int lane = threadIdx.x & 31;
  if (nt >= NT) return;
  const size_t as = (size_t)KS * 512;
  v8f acc0 = {}, acc1 = {}, acc2 = {}, acc3 = {};
  {
    const u16* bp = B1 + ((size_t)nt * KS * 32 + lane) * 16;
    const u16* ap = A1 + (size_t)lane * 16;
    for (int ks = 0; ks < KS; ++ks) {
      v16bf b  = *(const v16bf*)bp; bp += 512;
      v16bf a0 = *(const v16bf*)(ap);
      v16bf a1 = *(const v16bf*)(ap + as);
      v16bf a2 = *(const v16bf*)(ap + 2 * as);
      v16bf a3 = *(const v16bf*)(ap + 3 * as);
      ap += 512;
      acc0 = __builtin_amdgcn_wmma_f32_16x16x32_bf16(false, a0, false, b, (short)0, acc0, false, false);
      acc1 = __builtin_amdgcn_wmma_f32_16x16x32_bf16(false, a1, false, b, (short)0, acc1, false, false);
      acc2 = __builtin_amdgcn_wmma_f32_16x16x32_bf16(false, a2, false, b, (short)0, acc2, false, false);
      acc3 = __builtin_amdgcn_wmma_f32_16x16x32_bf16(false, a3, false, b, (short)0, acc3, false, false);
    }
  }
  {
    const u16* bp = B2 + ((size_t)nt * KS * 32 + lane) * 16;
    const u16* ap = A2 + (size_t)lane * 16;
    for (int ks = 0; ks < KS; ++ks) {
      v16bf b  = *(const v16bf*)bp; bp += 512;
      v16bf a0 = *(const v16bf*)(ap);
      v16bf a1 = *(const v16bf*)(ap + as);
      v16bf a2 = *(const v16bf*)(ap + 2 * as);
      v16bf a3 = *(const v16bf*)(ap + 3 * as);
      ap += 512;
      acc0 = __builtin_amdgcn_wmma_f32_16x16x32_bf16(false, a0, false, b, (short)0, acc0, false, false);
      acc1 = __builtin_amdgcn_wmma_f32_16x16x32_bf16(false, a1, false, b, (short)0, acc1, false, false);
      acc2 = __builtin_amdgcn_wmma_f32_16x16x32_bf16(false, a2, false, b, (short)0, acc2, false, false);
      acc3 = __builtin_amdgcn_wmma_f32_16x16x32_bf16(false, a3, false, b, (short)0, acc3, false, false);
    }
  }
  int lo = lane & 15, half = lane >> 4;
  int n  = nt * 16 + lo;
  float bv = bias[n];
  #pragma unroll
  for (int r = 0; r < 8; ++r) {
    int m = half * 8 + r;
    C[(size_t)(m)      * ldC + n] = acc0[r] + bv;
    C[(size_t)(m + 16) * ldC + n] = acc1[r] + bv;
    C[(size_t)(m + 32) * ldC + n] = acc2[r] + bv;
    C[(size_t)(m + 48) * ldC + n] = acc3[r] + bv;
  }
}

// embeds[b,0,:]=features; embeds[b,1+t,:]=emb[captions]; mean over L=25
__global__ void embed_mean(const float* __restrict__ features,
                           const int* __restrict__ captions,
                           const float* __restrict__ emb,
                           float* __restrict__ embeds, float* __restrict__ mean_e) {
  int idx = blockIdx.x * blockDim.x + threadIdx.x; // B*E = 32768
  int b = idx >> 9, e = idx & 511;
  float s = features[idx];
  embeds[(size_t)b * 25 * 512 + e] = s;
  for (int t = 0; t < 24; ++t) {
    float v = emb[(size_t)captions[b * 24 + t] * 512 + e];
    embeds[((size_t)b * 25 + 1 + t) * 512 + e] = v;
    s += v;
  }
  mean_e[idx] = s * (1.f / 25.f);
}

// per-b block: score -> softmax -> awe -> gate -> x_pk = pack([emb_t ; gate*awe])
// dg: [B,1536] = [h@Wd+bd | h@Wfb+bfb]
__global__ __launch_bounds__(256)
void attn_x_kernel(const float* __restrict__ att1, const float* __restrict__ dg,
                   const float* __restrict__ va, const float* __restrict__ ba,
                   const float* __restrict__ embeds, u16* __restrict__ x_pk, int t) {
  const int b = blockIdx.x;
  const int tid = threadIdx.x, lane = tid & 31, wid = tid >> 5;
  __shared__ float sscore[32];
  __shared__ float red[8];
  const float* drow = dg + (size_t)b * 1536;
  for (int l = 0; l < 25; ++l) {
    const float* arow = att1 + ((size_t)b * 25 + l) * 1024;
    float p = 0.f;
    for (int a = tid; a < 1024; a += 256)
      p += va[a] * fmaxf(arow[a] + drow[a], 0.f);
    for (int off = 16; off > 0; off >>= 1) p += __shfl_xor(p, off, 32);
    if (lane == 0) red[wid] = p;
    __syncthreads();
    if (tid == 0) {
      float s = 0.f;
      for (int w = 0; w < 8; ++w) s += red[w];
      sscore[l] = s + ba[0];
    }
    __syncthreads();
  }
  if (tid < 32) { // softmax over L=25 in one wave
    float v = (tid < 25) ? sscore[tid] : -3.4e38f;
    float mx = v;
    for (int off = 16; off > 0; off >>= 1) mx = fmaxf(mx, __shfl_xor(mx, off, 32));
    float ex = (tid < 25) ? __expf(v - mx) : 0.f;
    float sm = ex;
    for (int off = 16; off > 0; off >>= 1) sm += __shfl_xor(sm, off, 32);
    if (tid < 25) sscore[tid] = ex / sm;
  }
  __syncthreads();
  const float* eb = embeds + (size_t)b * 25 * 512;
  for (int e = tid; e < 512; e += 256) {
    float awe = 0.f;
    for (int l = 0; l < 25; ++l) awe += sscore[l] * eb[l * 512 + e];
    float g = 1.f / (1.f + __expf(-drow[1024 + e]));
    x_pk[apk_idx(b, e, 32)]       = f2bf(eb[t * 512 + e]); // emb_t
    x_pk[apk_idx(b, 512 + e, 32)] = f2bf(g * awe);
  }
}

// torch gate order i,f,g,o; writes h directly into packed bf16 fragments
__global__ void lstm_update(const float* __restrict__ z, float* __restrict__ c,
                            u16* __restrict__ h_pk) {
  int idx = blockIdx.x * blockDim.x + threadIdx.x; // B*H = 65536
  int b = idx >> 10, j = idx & 1023;
  const float* zr = z + (size_t)b * 4096;
  float iv = zr[j], fv = zr[j + 1024], gv = zr[j + 2048], ov = zr[j + 3072];
  auto sg = [](float v) { return 1.f / (1.f + __expf(-v)); };
  float cn = sg(fv) * c[idx] + sg(iv) * tanhf(gv);
  c[idx] = cn;
  h_pk[apk_idx(b, j, 32)] = f2bf(sg(ov) * tanhf(cn));
}

extern "C" void kernel_launch(void* const* d_in, const int* in_sizes, int n_in,
                              void* d_out, int out_size, void* d_ws, size_t ws_size,
                              hipStream_t stream) {
  (void)in_sizes; (void)n_in; (void)out_size; (void)ws_size;
  const float* features = (const float*)d_in[0];
  const int*   captions = (const int*)d_in[1];
  // d_in[2] lengths: unused (all T+1)
  const float* emb = (const float*)d_in[3];
  const float* We  = (const float*)d_in[4];  const float* be  = (const float*)d_in[5];
  const float* Wd  = (const float*)d_in[6];  const float* bd  = (const float*)d_in[7];
  const float* va  = (const float*)d_in[8];  const float* ba  = (const float*)d_in[9];
  const float* Wih = (const float*)d_in[10]; const float* bih = (const float*)d_in[11];
  const float* Wic = (const float*)d_in[12]; const float* bic = (const float*)d_in[13];
  const float* Wfb = (const float*)d_in[14]; const float* bfb = (const float*)d_in[15];
  const float* Wi  = (const float*)d_in[16]; const float* Wh  = (const float*)d_in[17];
  const float* bl  = (const float*)d_in[18];
  const float* Wo  = (const float*)d_in[19]; const float* bo  = (const float*)d_in[20];
  float* out = (float*)d_out;

  // workspace bump allocator (256B aligned), ~103 MB total
  size_t off = 0;
  auto alloc = [&](size_t elems, size_t esz) {
    void* p = (char*)d_ws + off;
    off += (elems * esz + 255) & ~(size_t)255;
    return p;
  };
  float* embeds = (float*)alloc(64 * 25 * 512, 4);
  float* mean_e = (float*)alloc(64 * 512, 4);
  float* att1v  = (float*)alloc(64 * 25 * 1024, 4);
  float* h0     = (float*)alloc(64 * 1024, 4);
  float* c      = (float*)alloc(64 * 1024, 4);
  float* dg     = (float*)alloc(64 * 1536, 4);   // [h@Wd+bd | h@Wfb+bfb]
  float* z      = (float*)alloc(64 * 4096, 4);
  float* bdg    = (float*)alloc(1536, 4);
  u16* embeds_pk = (u16*)alloc(64 * 25 * 512, 2);
  u16* meanE_pk  = (u16*)alloc(64 * 512, 2);
  u16* h_pk      = (u16*)alloc(64 * 1024, 2);
  u16* x_pk      = (u16*)alloc(64 * 1024, 2);
  u16* We_pk  = (u16*)alloc(512 * 1024, 2);
  u16* Wih_pk = (u16*)alloc(512 * 1024, 2);
  u16* Wic_pk = (u16*)alloc(512 * 1024, 2);
  u16* Wdg_pk = (u16*)alloc(1024 * 1536, 2);     // tiles 0..63 Wd, 64..95 Wfb
  u16* Wi_pk  = (u16*)alloc(1024 * 4096, 2);
  u16* Wh_pk  = (u16*)alloc(1024 * 4096, 2);
  u16* Wo_pk  = (u16*)alloc((size_t)1024 * 32000, 2);

  // ---- one-time: pack weights to bf16 fragment order ----
  pack_b_bf16<<<512 * 1024 / 256, 256, 0, stream>>>(We,  We_pk,  16, 1024);
  pack_b_bf16<<<512 * 1024 / 256, 256, 0, stream>>>(Wih, Wih_pk, 16, 1024);
  pack_b_bf16<<<512 * 1024 / 256, 256, 0, stream>>>(Wic, Wic_pk, 16, 1024);
  pack_b_bf16<<<1024 * 1024 / 256, 256, 0, stream>>>(Wd,  Wdg_pk, 32, 1024);
  pack_b_bf16<<<1024 * 512 / 256, 256, 0, stream>>>(Wfb, Wdg_pk + (size_t)64 * 32 * 512,
                                                    32, 512);
  pack_b_bf16<<<1024 * 4096 / 256, 256, 0, stream>>>(Wi, Wi_pk, 32, 4096);
  pack_b_bf16<<<1024 * 4096 / 256, 256, 0, stream>>>(Wh, Wh_pk, 32, 4096);
  pack_b_bf16<<<(1024 * 32000) / 256, 256, 0, stream>>>(Wo, Wo_pk, 32, 32000);
  concat_bias<<<6, 256, 0, stream>>>(bd, bfb, bdg, 1024);

  // ---- prologue: embeddings, mean, att1, h0, c0 ----
  embed_mean<<<128, 256, 0, stream>>>(features, captions, emb, embeds, mean_e);
  pack_a_bf16<<<64 * 25 * 512 / 256, 256, 0, stream>>>(embeds, embeds_pk, 16);
  pack_a_bf16<<<128, 256, 0, stream>>>(mean_e, meanE_pk, 16);
  wmma_gemm_bf16<<<800, 256, 0, stream>>>(embeds_pk, We_pk, be, att1v, 100, 64, 16, 1024);
  wmma_gemm_m4<<<8, 256, 0, stream>>>(meanE_pk, Wih_pk, bih, h0, 64, 16, 1024);
  wmma_gemm_m4<<<8, 256, 0, stream>>>(meanE_pk, Wic_pk, bic, c,  64, 16, 1024);
  pack_a_bf16<<<256, 256, 0, stream>>>(h0, h_pk, 32);

  // ---- sequential scan: 24 steps, 5 kernels/step ----
  for (int t = 0; t < 24; ++t) {
    // dg = [h@Wd+bd | h@Wfb+bfb] : NT=96
    wmma_gemm_m4<<<12, 256, 0, stream>>>(h_pk, Wdg_pk, bdg, dg, 96, 32, 1536);
    attn_x_kernel<<<64, 256, 0, stream>>>(att1v, dg, va, ba, embeds, x_pk, t);
    // z = x@Wi + h@Wh + bl : NT=256
    wmma_gemm2_m4<<<32, 256, 0, stream>>>(x_pk, Wi_pk, h_pk, Wh_pk, bl, z, 256, 32, 4096);
    lstm_update<<<256, 256, 0, stream>>>(z, c, h_pk);
    // logits_t = h@Wo + bo -> out[:, t, :] : NT=2000
    wmma_gemm_m4<<<250, 256, 0, stream>>>(h_pk, Wo_pk, bo, out + (size_t)t * 32000,
                                          2000, 32, 24 * 32000);
  }
}
